// BKTransformer_Origin_21208548508272
// MI455X (gfx1250) — compile-verified, matrix operand order checked by reading, not run on hardware
//
#include <hip/hip_runtime.h>
#include <hip/hip_bf16.h>
#include <math.h>

// ---------------------------------------------------------------------------
// BKTransformer forward for MI455X (gfx1250, wave32, WMMA).
// GEMMs use v_wmma_f32_16x16x32_f16; attention is flash-style (no BxHxTxT
// score tensor); sequential BKT scan runs 16 blocks x 128 threads.
// ---------------------------------------------------------------------------

typedef __attribute__((ext_vector_type(16))) _Float16 v16h;
typedef __attribute__((ext_vector_type(8)))  float    v8f;
typedef __attribute__((ext_vector_type(4)))  unsigned int u32x4;
typedef __attribute__((ext_vector_type(4)))  float    f32x4;

union FragU  { v16h v; u32x4 u[2]; };   // 32B: one WMMA f16 A/B fragment
union F16x8  { u32x4 u; _Float16 h[8]; };

static constexpr int kB  = 16;
static constexpr int kT  = 512;
static constexpr int kE  = 512;
static constexpr int kH  = 8;
static constexpr int kDH = 64;
static constexpr int kNS = 128;
static constexpr int kM  = kB * kT;          // 8192 tokens

static __device__ __forceinline__ float wsum16(float v) {
  v += __shfl_xor(v, 1);  v += __shfl_xor(v, 2);
  v += __shfl_xor(v, 4);  v += __shfl_xor(v, 8);
  return v;
}
static __device__ __forceinline__ float wmax16(float v) {
  v = fmaxf(v, __shfl_xor(v, 1)); v = fmaxf(v, __shfl_xor(v, 2));
  v = fmaxf(v, __shfl_xor(v, 4)); v = fmaxf(v, __shfl_xor(v, 8));
  return v;
}
static __device__ __forceinline__ float wsum32(float v) {
  v = wsum16(v); v += __shfl_xor(v, 16); return v;
}
static __device__ __forceinline__ float sigmoidf(float x) {
  return 1.f / (1.f + __expf(-x));
}

// ---------------------------------------------------------------------------
// Generic WMMA GEMM:  out = act(A[M,K]f16 * W[N,K]^T f16 + bias) (+ resid)
// Block = 256 threads (8 waves), tile 128x128, K-chunk 32.
// ACT: 0 none, 1 exact GELU, 2 ReLU, 3 PReLU(alpha)
// ---------------------------------------------------------------------------
template<int ACT>
__global__ __launch_bounds__(256)
void gemm_wmma(const _Float16* __restrict__ A, const _Float16* __restrict__ W,
               const float* __restrict__ bias, const float* __restrict__ resid,
               float* __restrict__ outF, _Float16* __restrict__ outH,
               int M, int N, int K, const float* __restrict__ alphaPtr)
{
  __shared__ _Float16 As[128 * 32];
  __shared__ _Float16 Bs[128 * 32];

  const int tid  = threadIdx.x;
  const int lane = tid & 31;
  const int w    = tid >> 5;
  const int wm   = w & 3;            // 4 wave rows
  const int wn   = w >> 2;           // 2 wave cols
  const int bm   = blockIdx.y * 128;
  const int bn   = blockIdx.x * 128;
  const int hi8  = (lane & 16) ? 8 : 0;   // K-half / M-half selector
  const int l15  = lane & 15;
  const float alpha = alphaPtr ? *alphaPtr : 0.f;

  v8f acc[2][4];
  const v8f vz = {0.f,0.f,0.f,0.f,0.f,0.f,0.f,0.f};
  #pragma unroll
  for (int mi = 0; mi < 2; ++mi)
    #pragma unroll
    for (int ni = 0; ni < 4; ++ni) acc[mi][ni] = vz;

  const int q0 = tid, q1 = tid + 256;   // 512 u32x4 per 128x32 tile
  for (int k0 = 0; k0 < K; k0 += 32) {
    {
      const int r0 = q0 >> 2, c0 = (q0 & 3) * 8;
      const int r1 = q1 >> 2, c1 = (q1 & 3) * 8;
      ((u32x4*)As)[q0] = *(const u32x4*)(A + (size_t)(bm + r0) * K + k0 + c0);
      ((u32x4*)As)[q1] = *(const u32x4*)(A + (size_t)(bm + r1) * K + k0 + c1);
      ((u32x4*)Bs)[q0] = *(const u32x4*)(W + (size_t)(bn + r0) * K + k0 + c0);
      ((u32x4*)Bs)[q1] = *(const u32x4*)(W + (size_t)(bn + r1) * K + k0 + c1);
      if (k0 + 32 < K) {
        __builtin_prefetch(A + (size_t)(bm + r0) * K + k0 + 32 + c0, 0, 1);
        __builtin_prefetch(W + (size_t)(bn + r0) * K + k0 + 32 + c0, 0, 1);
      }
    }
    __syncthreads();

    v16h af[2], bf[4];
    #pragma unroll
    for (int mi = 0; mi < 2; ++mi) {
      const _Float16* rp = As + (wm * 32 + mi * 16 + l15) * 32;
      FragU f; f.u[0] = *(const u32x4*)(rp + hi8); f.u[1] = *(const u32x4*)(rp + hi8 + 16);
      af[mi] = f.v;
    }
    #pragma unroll
    for (int ni = 0; ni < 4; ++ni) {
      const _Float16* rp = Bs + (wn * 64 + ni * 16 + l15) * 32;
      FragU f; f.u[0] = *(const u32x4*)(rp + hi8); f.u[1] = *(const u32x4*)(rp + hi8 + 16);
      bf[ni] = f.v;
    }
    #pragma unroll
    for (int mi = 0; mi < 2; ++mi)
      #pragma unroll
      for (int ni = 0; ni < 4; ++ni)
        acc[mi][ni] = __builtin_amdgcn_wmma_f32_16x16x32_f16(
            false, af[mi], false, bf[ni], (short)0, acc[mi][ni], false, false);
    __syncthreads();
  }

  #pragma unroll
  for (int mi = 0; mi < 2; ++mi) {
    const int row = bm + wm * 32 + mi * 16 + hi8;
    #pragma unroll
    for (int ni = 0; ni < 4; ++ni) {
      const int col = bn + wn * 64 + ni * 16 + l15;
      const float bc = bias[col];
      #pragma unroll
      for (int r = 0; r < 8; ++r) {
        float v = acc[mi][ni][r] + bc;
        if (ACT == 1)      v = 0.5f * v * (1.f + erff(v * 0.70710678118f));
        else if (ACT == 2) v = v > 0.f ? v : 0.f;
        else if (ACT == 3) v = v >= 0.f ? v : alpha * v;
        const size_t o = (size_t)(row + r) * N + col;
        if (outF) outF[o] = (resid ? resid[o] : 0.f) + v;
        if (outH) outH[o] = (_Float16)v;
      }
    }
  }
}

// ---------------------------------------------------------------------------
// Flash attention: one wave per 16-row Q tile, key tiles of 32, WMMA for
// S = QK^T and O += P V.  qkv is [M, 3E] f16 (q|k|v), out is [M, E] f16.
// ---------------------------------------------------------------------------
__global__ __launch_bounds__(256)
void attn_kernel(const _Float16* __restrict__ qkv, _Float16* __restrict__ outh)
{
  __shared__ _Float16 Pl[8][16 * 32];   // per-wave P tile   (1 KB each)
  __shared__ _Float16 Vt[8][64 * 32];   // per-wave V^T tile (4 KB each)

  const int tid  = threadIdx.x;
  const int lane = tid & 31;
  const int w    = tid >> 5;
  const int qt   = blockIdx.x * 8 + w;      // global q-tile id
  const int b    = qt >> 8;                 // 256 tiles per batch (H * T/16)
  const int rem  = qt & 255;
  const int h    = rem >> 5;
  const int q16  = rem & 31;                // q-tile within sequence
  const int hi8  = (lane & 16) ? 8 : 0;
  const int l15  = lane & 15;
  const int S3E  = 3 * kE;

  // Q fragments (16 x 64, two K-chunks of 32)
  v16h qa[2];
  {
    const _Float16* qrow = qkv + (size_t)(b * kT + q16 * 16 + l15) * S3E + h * kDH;
    #pragma unroll
    for (int c = 0; c < 2; ++c) {
      FragU f;
      f.u[0] = *(const u32x4*)(qrow + c * 32 + hi8);
      f.u[1] = *(const u32x4*)(qrow + c * 32 + hi8 + 16);
      qa[c] = f.v;
    }
  }

  const v8f vz = {0.f,0.f,0.f,0.f,0.f,0.f,0.f,0.f};
  v8f accO[4];
  #pragma unroll
  for (int c = 0; c < 4; ++c) accO[c] = vz;
  float rowM[8], rowL[8];
  #pragma unroll
  for (int r = 0; r < 8; ++r) { rowM[r] = -1e30f; rowL[r] = 0.f; }

  const int ktMax = q16 >> 1;           // causal: keys <= q16*16+15
  for (int kt = 0; kt <= ktMax; ++kt) {
    // ---- S = Q K^T  (16 x 32 scores, 4 WMMAs) ----
    v8f sac[2]; sac[0] = vz; sac[1] = vz;
    #pragma unroll
    for (int s = 0; s < 2; ++s) {
      const _Float16* krow = qkv + (size_t)(b * kT + kt * 32 + s * 16 + l15) * S3E + kE + h * kDH;
      #pragma unroll
      for (int c = 0; c < 2; ++c) {
        FragU f;
        f.u[0] = *(const u32x4*)(krow + c * 32 + hi8);
        f.u[1] = *(const u32x4*)(krow + c * 32 + hi8 + 16);
        sac[s] = __builtin_amdgcn_wmma_f32_16x16x32_f16(
            false, qa[c], false, f.v, (short)0, sac[s], false, false);
      }
    }
    // ---- stage V tile transposed into LDS (lane = key) ----
    {
      const _Float16* vrow = qkv + (size_t)(b * kT + kt * 32 + lane) * S3E + 2 * kE + h * kDH;
      #pragma unroll
      for (int d8 = 0; d8 < 8; ++d8) {
        F16x8 ch; ch.u = *(const u32x4*)(vrow + d8 * 8);
        #pragma unroll
        for (int e = 0; e < 8; ++e) Vt[w][(d8 * 8 + e) * 32 + lane] = ch.h[e];
      }
    }
    // ---- scale + causal mask ----
    float fs[2][8];
    #pragma unroll
    for (int s = 0; s < 2; ++s)
      #pragma unroll
      for (int r = 0; r < 8; ++r) {
        const int qpos = q16 * 16 + r + hi8;
        const int key  = kt * 32 + s * 16 + l15;
        const float v  = sac[s][r] * 0.125f;           // 1/sqrt(64)
        fs[s][r] = (key > qpos) ? -1e30f : v;
      }
    // ---- online softmax (rows live in 16-lane groups) ----
    float pe[2][8];
    #pragma unroll
    for (int r = 0; r < 8; ++r) {
      float mx = wmax16(fmaxf(fs[0][r], fs[1][r]));
      const float nm = fmaxf(rowM[r], mx);
      const float al = __expf(rowM[r] - nm);
      const float p0 = __expf(fs[0][r] - nm);
      const float p1 = __expf(fs[1][r] - nm);
      pe[0][r] = p0; pe[1][r] = p1;
      rowL[r] = rowL[r] * al + wsum16(p0 + p1);
      rowM[r] = nm;
      #pragma unroll
      for (int c = 0; c < 4; ++c) accO[c][r] *= al;
    }
    // ---- P -> LDS (C-layout to A-layout requires cross-lane move) ----
    #pragma unroll
    for (int s = 0; s < 2; ++s)
      #pragma unroll
      for (int r = 0; r < 8; ++r)
        Pl[w][(r + hi8) * 32 + s * 16 + l15] = (_Float16)pe[s][r];
    asm volatile("s_wait_dscnt 0" ::: "memory");   // per-wave LDS RAW fence
    // ---- O += P V  (4 WMMAs over DH chunks) ----
    v16h pf;
    {
      const _Float16* rp = &Pl[w][0] + l15 * 32;
      FragU f; f.u[0] = *(const u32x4*)(rp + hi8); f.u[1] = *(const u32x4*)(rp + hi8 + 16);
      pf = f.v;
    }
    #pragma unroll
    for (int c = 0; c < 4; ++c) {
      const _Float16* rp = &Vt[w][0] + (c * 16 + l15) * 32;
      FragU f; f.u[0] = *(const u32x4*)(rp + hi8); f.u[1] = *(const u32x4*)(rp + hi8 + 16);
      accO[c] = __builtin_amdgcn_wmma_f32_16x16x32_f16(
          false, pf, false, f.v, (short)0, accO[c], false, false);
    }
  }

  #pragma unroll
  for (int c = 0; c < 4; ++c)
    #pragma unroll
    for (int r = 0; r < 8; ++r) {
      const int tok = b * kT + q16 * 16 + r + hi8;
      const int col = h * kDH + c * 16 + l15;
      outh[(size_t)tok * kE + col] = (_Float16)(accO[c][r] / rowL[r]);
    }
}

// ---------------------------------------------------------------------------
// LayerNorm: one wave per row of E=512, f32 in -> f16 out.
// ---------------------------------------------------------------------------
__global__ __launch_bounds__(256)
void ln_kernel(const float* __restrict__ x, const float* __restrict__ g,
               const float* __restrict__ bt, _Float16* __restrict__ out)
{
  const int lane = threadIdx.x & 31;
  const int row  = blockIdx.x * 8 + (threadIdx.x >> 5);
  const float* xr = x + (size_t)row * kE;
  f32x4 v[4];
  float s = 0.f;
  #pragma unroll
  for (int i = 0; i < 4; ++i) {
    v[i] = ((const f32x4*)xr)[lane * 4 + i];
    s += v[i].x + v[i].y + v[i].z + v[i].w;
  }
  const float mean = wsum32(s) * (1.f / kE);
  float d = 0.f;
  #pragma unroll
  for (int i = 0; i < 4; ++i) {
    f32x4 t = v[i] - mean;
    d += t.x*t.x + t.y*t.y + t.z*t.z + t.w*t.w;
  }
  const float rinv = rsqrtf(wsum32(d) * (1.f / kE) + 1e-5f);
  F16x8 o[2];
  #pragma unroll
  for (int i = 0; i < 4; ++i) {
    const f32x4 gg = ((const f32x4*)g )[lane * 4 + i];
    const f32x4 bb = ((const f32x4*)bt)[lane * 4 + i];
    const f32x4 y  = (v[i] - mean) * rinv * gg + bb;
    o[i >> 1].h[(i & 1) * 4 + 0] = (_Float16)y.x;
    o[i >> 1].h[(i & 1) * 4 + 1] = (_Float16)y.y;
    o[i >> 1].h[(i & 1) * 4 + 2] = (_Float16)y.z;
    o[i >> 1].h[(i & 1) * 4 + 3] = (_Float16)y.w;
  }
  u32x4* op = (u32x4*)(out + (size_t)row * kE + lane * 16);
  op[0] = o[0].u; op[1] = o[1].u;
}

// ---------------------------------------------------------------------------
// Small kernels
// ---------------------------------------------------------------------------
__global__ void cvt_kernel(const float* __restrict__ s, _Float16* __restrict__ d, int n)
{
  const int i = blockIdx.x * 256 + threadIdx.x;
  if (i < n) d[i] = (_Float16)s[i];
}

__global__ void embed_kernel(const int* __restrict__ obs, const float* __restrict__ se,
                             const float* __restrict__ ce, const float* __restrict__ pe,
                             float* __restrict__ x)
{
  const int idx = blockIdx.x * 256 + threadIdx.x;   // M*E
  const int tok = idx >> 9, e = idx & 511;
  const int sid = obs[tok * 2];
  int cr = obs[tok * 2 + 1]; cr = cr < 0 ? 0 : (cr > 1 ? 1 : cr);
  const int t = tok & (kT - 1);
  x[idx] = se[sid * kE + e] + ce[cr * kE + e] + pe[t * kE + e];
}

__global__ void zero_kernel(float* __restrict__ a)
{
  if (threadIdx.x < 16) a[threadIdx.x] = 0.f;
}

__global__ void spwo_kernel(const _Float16* __restrict__ z, const float* __restrict__ wo,
                            const float* __restrict__ bo, float* __restrict__ logits)
{
  const int s = threadIdx.x;          // 128 skills
  const _Float16* zr = z + (size_t)s * kE;
  for (int j = 0; j < 5; ++j) {
    float a = bo[j];
    const float* wr = wo + j * kE;
    for (int e = 0; e < kE; ++e) a += (float)zr[e] * wr[e];
    logits[s * 5 + j] = a;
  }
}

__global__ void opterms_kernel(const float* __restrict__ logits, float* __restrict__ acc)
{
  const int s = threadIdx.x;          // 128 threads, 4 waves
  const int lane = s & 31, w = s >> 5;
  float op[4];
  #pragma unroll
  for (int j = 0; j < 4; ++j) op[j] = sigmoidf(logits[s * 5 + 1 + j]);
  float r[3];
  r[0] = fmaxf(op[0] - (1.f - op[3]) / (op[2] + 1e-6f), 0.f);
  r[1] = fmaxf(op[2] - 0.5f, 0.f);
  r[2] = fmaxf(op[3] - 0.5f, 0.f);
  __shared__ float red[3][4];
  #pragma unroll
  for (int i = 0; i < 3; ++i) {
    float v = wsum32(r[i]);
    if (lane == 0) red[i][w] = v;
  }
  __syncthreads();
  if (s == 0)
    for (int i = 0; i < 3; ++i)
      acc[5 + i] = red[i][0] + red[i][1] + red[i][2] + red[i][3];
}

__global__ __launch_bounds__(256)
void ploss_kernel(const float* __restrict__ ld, const float* __restrict__ logits,
                  float* __restrict__ pout, float* __restrict__ acc)
{
  const int idx = blockIdx.x * 256 + threadIdx.x;   // B*T*NS
  const int tok = idx >> 7, s = idx & 127;
  const int lane = threadIdx.x & 31, w = threadIdx.x >> 5;
  float pv[4], l2 = 0.f;
  #pragma unroll
  for (int j = 0; j < 4; ++j) {
    const float d = ld[(size_t)tok * 512 + s * 4 + j];
    l2 += d * d;
    const float v = sigmoidf(logits[s * 5 + 1 + j] + d);
    pv[j] = v;
    pout[(size_t)idx * 4 + j] = v;
  }
  float vals[4];
  vals[0] = fmaxf(pv[0] - (1.f - pv[3]) / (pv[2] + 1e-6f), 0.f);
  vals[1] = fmaxf(pv[2] - 0.5f, 0.f);
  vals[2] = fmaxf(pv[3] - 0.5f, 0.f);
  vals[3] = l2;
  __shared__ float red[4][8];
  #pragma unroll
  for (int i = 0; i < 4; ++i) {
    float v = wsum32(vals[i]);
    if (lane == 0) red[i][w] = v;
  }
  __syncthreads();
  if (threadIdx.x < 4) {
    float ssum = 0.f;
    for (int k = 0; k < 8; ++k) ssum += red[threadIdx.x][k];
    atomicAdd(acc + threadIdx.x, ssum);
  }
}

__global__ __launch_bounds__(256)
void smooth_kernel(const float* __restrict__ ld, float* __restrict__ acc)
{
  const int idx = blockIdx.x * 256 + threadIdx.x;   // B*(T-1)*512
  const int b = idx / (511 * 512);
  const int r = idx - b * 511 * 512;
  const int t1 = r >> 9, col = r & 511;
  const size_t base = ((size_t)(b * kT) + t1) * 512 + col;
  const float d = ld[base + 512] - ld[base];
  float v = wsum32(d * d);
  __shared__ float red[8];
  const int lane = threadIdx.x & 31, w = threadIdx.x >> 5;
  if (lane == 0) red[w] = v;
  __syncthreads();
  if (threadIdx.x == 0) {
    float ssum = 0.f;
    for (int k = 0; k < 8; ++k) ssum += red[k];
    atomicAdd(acc + 4, ssum);
  }
}

// Sequential BKT scan: block = batch row, thread = skill.
__global__ __launch_bounds__(128)
void scan_kernel(const float* __restrict__ outp, const float* __restrict__ logits,
                 const float* __restrict__ p, float* __restrict__ corrects,
                 float* __restrict__ latents, float* __restrict__ acc)
{
  const int b = blockIdx.x, s = threadIdx.x;
  float lat = sigmoidf(logits[s * 5]);
  float bce = 0.f, cnt = 0.f;
  for (int t = 0; t < kT; ++t) {
    const float skf = outp[(size_t)(b * kT + t) * 2];
    const float tc  = outp[(size_t)(b * kT + t) * 2 + 1];
    const int sk = (skf == -1000.f) ? 0 : (int)skf;
    const float lc = fminf(fmaxf(lat, 1e-5f), 1.f - 1e-5f);
    const float* pp = p + ((size_t)(b * kT + t) * kNS + s) * 4;
    const float learning = pp[0], guess = pp[2], slip = pp[3];
    const float correct = lc * (1.f - slip) + (1.f - lc) * guess;
    latents[((size_t)t * kB + b) * kNS + s] = lc;
    float nl = lc;
    if (s == sk) {
      corrects[b * kT + t] = correct;
      const float a1v = lc * (1.f - slip), a0 = (1.f - lc) * guess;
      const float kt1 = a1v / (a1v + a0);
      const float b1v = lc * slip, b0 = (1.f - lc) * (1.f - guess);
      const float kt0 = b1v / (b1v + b0);
      const float sel = (tc > 0.5f) ? kt1 : kt0;
      nl = sel + (1.f - sel) * learning;
      if (tc != -1000.f) {
        const float c = fminf(fmaxf(correct, 1e-7f), 1.f - 1e-7f);
        bce += -(tc * logf(c) + (1.f - tc) * logf(1.f - c));
      }
    }
    if (s == 0 && tc != -1000.f) cnt += 1.f;
    lat = fminf(fmaxf(nl, 1e-4f), 1.f - 1e-4f);
  }
  __shared__ float red[4];
  const int lane = s & 31, w = s >> 5;
  float v = wsum32(bce);
  if (lane == 0) red[w] = v;
  __syncthreads();
  if (s == 0) {
    atomicAdd(acc + 8, red[0] + red[1] + red[2] + red[3]);
    atomicAdd(acc + 9, cnt);
  }
}

__global__ void finalize_kernel(const float* __restrict__ acc, float* __restrict__ lossOut)
{
  const float inv_btns = 1.f / (16.f * 512.f * 128.f);
  const float inv_ns   = 1.f / 128.f;
  const float m1 = acc[0] * inv_btns + acc[5] * inv_ns;
  const float m2 = acc[1] * inv_btns + acc[6] * inv_ns;
  const float m3 = acc[2] * inv_btns + acc[7] * inv_ns;
  float loss = 50.f * (m1 + m2 + m3)
             + acc[3] / (16.f * 512.f * 512.f)
             + acc[4] / (16.f * 511.f * 512.f)
             + acc[8] / fmaxf(acc[9], 1.f);
  *lossOut = loss;
}

// ---------------------------------------------------------------------------
// Host driver
// ---------------------------------------------------------------------------
extern "C" void kernel_launch(void* const* d_in, const int* in_sizes, int n_in,
                              void* d_out, int out_size, void* d_ws, size_t ws_size,
                              hipStream_t stream)
{
  (void)in_sizes; (void)n_in; (void)out_size; (void)ws_size;

  // ---- input indices (setup_inputs dict order, depth-first) ----
  const int* obs      = (const int*)  d_in[0];
  const float* outp   = (const float*)d_in[1];
  const float* se     = (const float*)d_in[2];
  const float* ce     = (const float*)d_in[3];
  const float* pe     = (const float*)d_in[4];
  const int LBASE = 5, LSTR = 12;
  const float* lnfg = (const float*)d_in[53];
  const float* lnfb = (const float*)d_in[54];
  const float* hw1  = (const float*)d_in[55];
  const float* hb1  = (const float*)d_in[56];
  const float* a1p  = (const float*)d_in[57];
  const float* hw2  = (const float*)d_in[58];
  const float* hb2  = (const float*)d_in[59];
  const float* a2p  = (const float*)d_in[60];
  const float* hw3  = (const float*)d_in[61];
  const float* hb3  = (const float*)d_in[62];
  const float* spw[4] = {(const float*)d_in[63], (const float*)d_in[64],
                         (const float*)d_in[65], (const float*)d_in[66]};
  const float* spb[4] = {(const float*)d_in[67], (const float*)d_in[68],
                         (const float*)d_in[69], (const float*)d_in[70]};
  const float* spwo = (const float*)d_in[71];
  const float* spbo = (const float*)d_in[72];

  // ---- workspace layout ----
  unsigned char* base = (unsigned char*)d_ws;
  size_t off = 0;
  auto wsAlloc = [&](size_t bytes) -> void* {
    void* p = base + off;
    off += (bytes + 255) & ~(size_t)255;
    return p;
  };
  float*     acc   = (float*)    wsAlloc(64 * sizeof(float));
  _Float16*  wpool = (_Float16*) wsAlloc((size_t)14417920 * 2);
  float*     X     = (float*)    wsAlloc((size_t)kM * kE * 4);
  _Float16*  bufA  = (_Float16*) wsAlloc((size_t)kM * kE * 2);
  _Float16*  bufB  = (_Float16*) wsAlloc((size_t)kM * 4 * kE * 2);
  _Float16*  bufC  = (_Float16*) wsAlloc((size_t)kM * kE * 2);
  float*     ldiff = (float*)    wsAlloc((size_t)kM * 512 * 4);
  _Float16*  zb1   = (_Float16*) wsAlloc((size_t)kNS * kE * 2);
  _Float16*  zb2   = (_Float16*) wsAlloc((size_t)kNS * kE * 2);
  float*     logit = (float*)    wsAlloc((size_t)kNS * 5 * 4);

  size_t woff = 0;
  auto wslot = [&](size_t n) -> _Float16* { _Float16* p = wpool + woff; woff += n; return p; };
  _Float16 *wqkv_h[4], *wo_h[4], *w1_h[4], *w2_h[4];
  for (int l = 0; l < 4; ++l) {
    wqkv_h[l] = wslot((size_t)1536 * 512);
    wo_h[l]   = wslot((size_t)512 * 512);
    w1_h[l]   = wslot((size_t)2048 * 512);
    w2_h[l]   = wslot((size_t)512 * 2048);
  }
  _Float16* hw1_h = wslot((size_t)512 * 512);
  _Float16* hw2_h = wslot((size_t)512 * 512);
  _Float16* hw3_h = wslot((size_t)512 * 512);
  _Float16* spw_h[4];
  for (int i = 0; i < 4; ++i) spw_h[i] = wslot((size_t)512 * 512);

  // ---- output layout (tuple: corrects, latents, p, loss) ----
  float* out     = (float*)d_out;
  float* o_corr  = out;
  float* o_lat   = out + (size_t)kB * kT;
  float* o_p     = o_lat + (size_t)kT * kB * kNS;
  float* o_loss  = o_p + (size_t)kB * kT * kNS * 4;

  auto cvt = [&](const float* s, _Float16* d, int n) {
    cvt_kernel<<<dim3((n + 255) / 256), dim3(256), 0, stream>>>(s, d, n);
  };
  auto gemm = [&](int act, const _Float16* A, const _Float16* W, const float* bias,
                  const float* resid, float* oF, _Float16* oH,
                  int M, int N, int K, const float* alpha) {
    dim3 g(N / 128, M / 128), blk(256);
    switch (act) {
      case 0: gemm_wmma<0><<<g, blk, 0, stream>>>(A, W, bias, resid, oF, oH, M, N, K, alpha); break;
      case 1: gemm_wmma<1><<<g, blk, 0, stream>>>(A, W, bias, resid, oF, oH, M, N, K, alpha); break;
      case 2: gemm_wmma<2><<<g, blk, 0, stream>>>(A, W, bias, resid, oF, oH, M, N, K, alpha); break;
      default:gemm_wmma<3><<<g, blk, 0, stream>>>(A, W, bias, resid, oF, oH, M, N, K, alpha); break;
    }
  };

  // 0) zero loss accumulators
  zero_kernel<<<dim3(1), dim3(64), 0, stream>>>(acc);

  // 1) convert all GEMM weights to f16 once
  for (int l = 0; l < 4; ++l) {
    cvt((const float*)d_in[LBASE + LSTR * l + 2],  wqkv_h[l], 1536 * 512);
    cvt((const float*)d_in[LBASE + LSTR * l + 4],  wo_h[l],   512 * 512);
    cvt((const float*)d_in[LBASE + LSTR * l + 8],  w1_h[l],   2048 * 512);
    cvt((const float*)d_in[LBASE + LSTR * l + 10], w2_h[l],   512 * 2048);
  }
  cvt(hw1, hw1_h, 512 * 512);
  cvt(hw2, hw2_h, 512 * 512);
  cvt(hw3, hw3_h, 512 * 512);
  for (int i = 0; i < 4; ++i) cvt(spw[i], spw_h[i], 512 * 512);
  cvt(se, zb1, kNS * kE);   // skill_emb f16 for the skill-MLP path

  // 2) embeddings -> X
  embed_kernel<<<dim3(kM * kE / 256), dim3(256), 0, stream>>>(obs, se, ce, pe, X);

  // 3) transformer layers
  for (int l = 0; l < 4; ++l) {
    const float* ln1g = (const float*)d_in[LBASE + LSTR * l + 0];
    const float* ln1b = (const float*)d_in[LBASE + LSTR * l + 1];
    const float* bqkv = (const float*)d_in[LBASE + LSTR * l + 3];
    const float* bo   = (const float*)d_in[LBASE + LSTR * l + 5];
    const float* ln2g = (const float*)d_in[LBASE + LSTR * l + 6];
    const float* ln2b = (const float*)d_in[LBASE + LSTR * l + 7];
    const float* b1   = (const float*)d_in[LBASE + LSTR * l + 9];
    const float* b2   = (const float*)d_in[LBASE + LSTR * l + 11];

    ln_kernel<<<dim3(kM / 8), dim3(256), 0, stream>>>(X, ln1g, ln1b, bufA);
    gemm(0, bufA, wqkv_h[l], bqkv, nullptr, nullptr, bufB, kM, 1536, 512, nullptr);
    attn_kernel<<<dim3(kB * kH * (kT / 16) / 8), dim3(256), 0, stream>>>(bufB, bufC);
    gemm(0, bufC, wo_h[l], bo, X, X, nullptr, kM, 512, 512, nullptr);   // x += o W_o^T
    ln_kernel<<<dim3(kM / 8), dim3(256), 0, stream>>>(X, ln2g, ln2b, bufA);
    gemm(1, bufA, w1_h[l], b1, nullptr, nullptr, bufB, kM, 2048, 512, nullptr);  // GELU
    gemm(0, bufB, w2_h[l], b2, X, X, nullptr, kM, 512, 2048, nullptr);  // x += ...
  }

  // 4) head
  ln_kernel<<<dim3(kM / 8), dim3(256), 0, stream>>>(X, lnfg, lnfb, bufA);
  gemm(3, bufA, hw1_h, hb1, nullptr, nullptr, bufC, kM, 512, 512, a1p);  // PReLU(a1)
  gemm(3, bufC, hw2_h, hb2, nullptr, nullptr, bufA, kM, 512, 512, a2p);  // PReLU(a2)
  gemm(0, bufA, hw3_h, hb3, nullptr, ldiff, nullptr, kM, 512, 512, nullptr);

  // 5) skill path: z = relu-MLP(skill_emb); logits = z sp_wo^T + sp_bo
  gemm(2, zb1, spw_h[0], spb[0], nullptr, nullptr, zb2, kNS, 512, 512, nullptr);
  gemm(2, zb2, spw_h[1], spb[1], nullptr, nullptr, zb1, kNS, 512, 512, nullptr);
  gemm(2, zb1, spw_h[2], spb[2], nullptr, nullptr, zb2, kNS, 512, 512, nullptr);
  gemm(2, zb2, spw_h[3], spb[3], nullptr, nullptr, zb1, kNS, 512, 512, nullptr);
  spwo_kernel<<<dim3(1), dim3(128), 0, stream>>>(zb1, spwo, spbo, logit);

  // 6) p + loss partials
  opterms_kernel<<<dim3(1), dim3(128), 0, stream>>>(logit, acc);
  ploss_kernel<<<dim3(kB * kT * kNS / 256), dim3(256), 0, stream>>>(ldiff, logit, o_p, acc);
  smooth_kernel<<<dim3(kB * (kT - 1) * 512 / 256), dim3(256), 0, stream>>>(ldiff, acc);

  // 7) sequential BKT scan + BCE
  scan_kernel<<<dim3(kB), dim3(128), 0, stream>>>(outp, logit, o_p, o_corr, o_lat, acc);

  // 8) final loss
  finalize_kernel<<<dim3(1), dim3(1), 0, stream>>>(acc, o_loss);
}
